// RecurrentWarpFusionAligner_57535381897314
// MI455X (gfx1250) — matrix operand order repaired and checked by reading, not compiled, not run on hardware
//
#include <hip/hip_runtime.h>
#include <stdint.h>

#define VTOT (128*128*16)   // 262144 voxels
#define EPS 1e-5f

typedef __attribute__((ext_vector_type(16))) __bf16 v16bf;
typedef __attribute__((ext_vector_type(8)))  float  v8f;
typedef unsigned int u32;

// ---------- bf16 helpers (native hardware convert) ----------
static __device__ __forceinline__ unsigned short f2bf(float f) {
  __bf16 h = (__bf16)f;
  return __builtin_bit_cast(unsigned short, h);
}
static __device__ __forceinline__ float bf2f(unsigned short u) {
  unsigned int x = ((unsigned int)u) << 16;
  return __builtin_bit_cast(float, x);
}
static __device__ __forceinline__ __bf16 bfbits(unsigned short u) {
  return __builtin_bit_cast(__bf16, u);
}
static __device__ __forceinline__ float silu_f(float f) {
  return f / (1.0f + __expf(-f));
}

static __device__ __forceinline__ v8f wmma_bf16(v16bf a, v16bf b, v8f c) {
  return __builtin_amdgcn_wmma_f32_16x16x32_bf16(false, a, false, b, (short)0, c,
                                                 false, false);
}

union frag16 { v16bf v; u32 d[8]; };

// ---------- WMMA fragment loaders (bf16 stored as ushort) ----------
// A: 16x32 (MxK), row-major lda (lda must be even). K-pairs are contiguous ->
// one aligned 32-bit load per pair. kmax must be a multiple of 16.
static __device__ __forceinline__ v16bf load_a(const unsigned short* A, int lda,
                                               int row0, int k0, int kmax) {
  int l = threadIdx.x & 31;
  const unsigned short* Ar = A + (row0 + (l & 15)) * lda;
  int hi = (l >> 4) * 16;
  frag16 r;
#pragma unroll
  for (int j = 0; j < 8; ++j) {
    int k = k0 + 2 * j + hi;
    r.d[j] = (k < kmax) ? *(const u32*)(Ar + k) : 0u;
  }
  return r.v;
}
// B: 32x16 (KxN), row-major ldb (K-pairs strided; scalar loads).
static __device__ __forceinline__ v16bf load_b(const unsigned short* B, int ldb,
                                               int k0, int col0, int kmax) {
  int l = threadIdx.x & 31;
  int n = col0 + (l & 15);
  int kh = (l >> 4) * 16;
  v16bf b;
#pragma unroll
  for (int j = 0; j < 8; ++j) {
    int k = k0 + kh + 2 * j;
    b[2 * j]     = bfbits((k     < kmax) ? B[k * ldb + n]       : (unsigned short)0);
    b[2 * j + 1] = bfbits((k + 1 < kmax) ? B[(k + 1) * ldb + n] : (unsigned short)0);
  }
  return b;
}
// B where matrix is stored [N][K] row-major (B[k,n] = W[n*ldw + k]); pair loads.
static __device__ __forceinline__ v16bf load_bT(const unsigned short* W, int ldw,
                                                int k0, int n0, int kmax) {
  int l = threadIdx.x & 31;
  const unsigned short* Wr = W + (n0 + (l & 15)) * ldw;
  int kh = (l >> 4) * 16;
  frag16 r;
#pragma unroll
  for (int j = 0; j < 8; ++j) {
    int k = k0 + kh + 2 * j;
    r.d[j] = (k < kmax) ? *(const u32*)(Wr + k) : 0u;
  }
  return r.v;
}

#define ZERO8 {0.f,0.f,0.f,0.f,0.f,0.f,0.f,0.f}

// ---------- weight conversion ----------
__global__ void k_cvt_bf16(const float* __restrict__ s, unsigned short* __restrict__ d, int n) {
  int i = blockIdx.x * blockDim.x + threadIdx.x;
  if (i < n) d[i] = f2bf(s[i]);
}
// stem weights [64][97] -> padded [64][128] (zeros beyond 97)
__global__ void k_cvt_stem(const float* __restrict__ s, unsigned short* __restrict__ d) {
  int i = blockIdx.x * blockDim.x + threadIdx.x;
  if (i >= 64 * 128) return;
  int r = i >> 7, k = i & 127;
  d[i] = (k < 97) ? f2bf(s[r * 97 + k]) : (unsigned short)0;
}
// w: [64][64][27] -> dst: [27][64][64]
__global__ void k_repack_conv(const float* __restrict__ w, unsigned short* __restrict__ dst) {
  int i = blockIdx.x * blockDim.x + threadIdx.x;
  if (i >= 64 * 64 * 27) return;
  int tap = i % 27;
  int ci = (i / 27) & 63;
  int co = i / (27 * 64);
  dst[(tap * 64 + co) * 64 + ci] = f2bf(w[i]);
}

// ---------- activation transpose: x f32 [64][V] -> xt bf16 [V][64] ----------
__global__ __launch_bounds__(256) void k_x2bf(const float* __restrict__ x,
                                              unsigned short* __restrict__ xt) {
  __shared__ unsigned short s[64][68];
  int v0 = blockIdx.x * 64;
  int tid = threadIdx.x;
  int vi = tid & 63;
#pragma unroll
  for (int cc = 0; cc < 64; cc += 4) {
    int c = cc + (tid >> 6);
    s[vi][c] = f2bf(x[c * VTOT + v0 + vi]);
  }
  __syncthreads();
#pragma unroll
  for (int t = 0; t < 16; ++t) {
    int idx = t * 256 + tid;
    int vv = idx >> 6, c = idx & 63;
    xt[(v0 + vv) * 64 + c] = s[vv][c];
  }
}

// ---------- stem GEMM: y[64,V] = stem_w(64x97) @ concat ----------
__global__ __launch_bounds__(256) void k_stem(
    const float* __restrict__ hw, const float* __restrict__ fpa,
    const float* __restrict__ fc, const float* __restrict__ dt,
    const unsigned short* __restrict__ wA, float* __restrict__ y) {
  int wave = threadIdx.x >> 5;
  int l = threadIdx.x & 31;
  int col0 = (blockIdx.x * 8 + wave) * 16;
  int n = l & 15, kh = (l >> 4) * 16, hi = (l >> 4) * 8;
  int v = col0 + n;
  v8f acc[4] = {ZERO8, ZERO8, ZERO8, ZERO8};
#pragma unroll
  for (int kt = 0; kt < 4; ++kt) {
    v16bf b;
#pragma unroll
    for (int j = 0; j < 8; ++j) {
#pragma unroll
      for (int e = 0; e < 2; ++e) {
        int k = kt * 32 + kh + 2 * j + e;
        float f;
        if (k < 32)       f = hw[k * VTOT + v];
        else if (k < 64)  f = fpa[(k - 32) * VTOT + v];
        else if (k < 96)  f = fc[(k - 64) * VTOT + v];
        else if (k == 96) f = dt[v];
        else              f = 0.f;
        b[2 * j + e] = bfbits(f2bf(f));
      }
    }
#pragma unroll
    for (int mt = 0; mt < 4; ++mt) {
      v16bf a = load_a(wA, 128, mt * 16, kt * 32, 128);  // padded weights
      acc[mt] = wmma_bf16(a, b, acc[mt]);
    }
  }
#pragma unroll
  for (int mt = 0; mt < 4; ++mt)
#pragma unroll
    for (int r = 0; r < 8; ++r) {
      int m = mt * 16 + r + hi;
      y[m * VTOT + col0 + n] = acc[mt][r];
    }
}

// ---------- GroupNorm stats (deterministic 2-stage) ----------
__global__ __launch_bounds__(256) void k_gn_part(const float* __restrict__ y,
                                                 float* __restrict__ partial) {
  int g = blockIdx.y, b = blockIdx.x;
  int NB = gridDim.x;
  const long n = 8L * VTOT;
  const float* p = y + (long)g * n;
  float s = 0.f, s2 = 0.f;
  for (long i = (long)b * blockDim.x + threadIdx.x; i < n; i += (long)NB * blockDim.x) {
    float f = p[i];
    s += f; s2 += f * f;
  }
  __shared__ float sh[512];
  sh[threadIdx.x] = s; sh[256 + threadIdx.x] = s2;
  __syncthreads();
  for (int off = 128; off; off >>= 1) {
    if ((int)threadIdx.x < off) {
      sh[threadIdx.x] += sh[threadIdx.x + off];
      sh[256 + threadIdx.x] += sh[256 + threadIdx.x + off];
    }
    __syncthreads();
  }
  if (threadIdx.x == 0) {
    partial[(g * NB + b) * 2] = sh[0];
    partial[(g * NB + b) * 2 + 1] = sh[256];
  }
}
__global__ void k_gn_final(const float* __restrict__ partial, float* __restrict__ stats, int NB) {
  int g = threadIdx.x;
  if (g >= 8) return;
  float s = 0.f, s2 = 0.f;
  for (int b = 0; b < NB; ++b) {
    s += partial[(g * NB + b) * 2];
    s2 += partial[(g * NB + b) * 2 + 1];
  }
  float cnt = 8.0f * (float)VTOT;
  float m = s / cnt;
  float var = s2 / cnt - m * m;
  stats[2 * g] = m;
  stats[2 * g + 1] = rsqrtf(var + EPS);
}

// ---------- x = res + silu(GN(y)) ----------
__global__ __launch_bounds__(256) void k_gn_act(
    const float* __restrict__ y, const float* __restrict__ res,
    const float* __restrict__ gamma, const float* __restrict__ beta,
    const float* __restrict__ stats, float* __restrict__ out) {
  int i = blockIdx.x * blockDim.x + threadIdx.x;
  if (i >= 64 * VTOT) return;
  int c = i / VTOT;
  int g = c >> 3;
  float f = (y[i] - stats[2 * g]) * stats[2 * g + 1] * gamma[c] + beta[c];
  f = silu_f(f);
  out[i] = (res ? res[i] : 0.f) + f;
}

// ---------- dilated conv 3x3x3 (implicit tap-wise GEMM, bf16 activations) ----------
__global__ __launch_bounds__(128) void k_conv(
    const unsigned short* __restrict__ xt, const unsigned short* __restrict__ wt,
    float* __restrict__ y, int dil) {
  int wave = threadIdx.x >> 5;
  int xy = blockIdx.x * 4 + wave;       // 0..16383
  int xi = xy >> 7, yi = xy & 127;
  int l = threadIdx.x & 31;
  int n = l & 15, kh = (l >> 4) * 16, hi = (l >> 4) * 8;
  int basev = xy * 16;
  v8f acc[4] = {ZERO8, ZERO8, ZERO8, ZERO8};
  for (int tap = 0; tap < 27; ++tap) {
    int dx = tap / 9 - 1, dy = (tap / 3) % 3 - 1, dz = tap % 3 - 1;
    int ix = xi + dx * dil, iy = yi + dy * dil;
    if ((unsigned)ix >= 128u || (unsigned)iy >= 128u) continue;
    int iz = n + dz * dil;
    bool zok = (unsigned)iz < 16u;
    int srcv = (ix * 128 + iy) * 16 + (zok ? iz : 0);
    const unsigned short* xr = xt + srcv * 64;
    const unsigned short* W = wt + tap * 64 * 64;
#pragma unroll
    for (int kt = 0; kt < 2; ++kt) {
      frag16 b;
#pragma unroll
      for (int j = 0; j < 8; ++j) {
        int k = kt * 32 + kh + 2 * j;
        b.d[j] = zok ? *(const u32*)(xr + k) : 0u;
      }
#pragma unroll
      for (int mt = 0; mt < 4; ++mt) {
        v16bf a = load_a(W, 64, mt * 16, kt * 32, 64);
        acc[mt] = wmma_bf16(a, b.v, acc[mt]);
      }
    }
  }
#pragma unroll
  for (int mt = 0; mt < 4; ++mt)
#pragma unroll
    for (int r = 0; r < 8; ++r) {
      int m = mt * 16 + r + hi;
      y[m * VTOT + basev + n] = acc[mt][r];
    }
}

// ---------- window attention (fused, 1 block = 1 window) ----------
static __device__ __forceinline__ int win_voxel(int w, int t, int sx, int sy, int sz) {
  int wz = w & 3, wy = (w >> 2) & 15, wx = w >> 6;
  int az = t & 3, ay = (t >> 2) & 7, ax = t >> 5;
  int X = ((wx * 8 + ax) + sx) & 127;
  int Y = ((wy * 8 + ay) + sy) & 127;
  int Z = ((wz * 4 + az) + sz) & 15;
  return (X * 128 + Y) * 16 + Z;
}

__global__ __launch_bounds__(256) void k_attn(
    float* __restrict__ x, const unsigned short* __restrict__ xt,
    const float* __restrict__ stats,
    const float* __restrict__ n1g, const float* __restrict__ n1b,
    const unsigned short* __restrict__ wqkv, const float* __restrict__ bqkv,
    const unsigned short* __restrict__ wproj, const float* __restrict__ bproj,
    int sx, int sy, int sz) {
  __shared__ unsigned short s_tok[256 * 64];  // gn'd tokens, later reused for attn out
  __shared__ unsigned short s_q[256 * 64];
  __shared__ unsigned short s_k[256 * 64];
  __shared__ unsigned short s_v[256 * 64];
  __shared__ unsigned short s_p[8][16 * 32];  // per-wave P staging

  const int w = blockIdx.x;
  const int wave = threadIdx.x >> 5;
  const int l = threadIdx.x & 31;
  const int hi = (l >> 4) * 8;
  const v8f z8 = ZERO8;

  // gather (bf16, contiguous) + GroupNorm(norm1)
  {
    int t = threadIdx.x;
    int v = win_voxel(w, t, sx, sy, sz);
    const unsigned short* xr = xt + v * 64;
#pragma unroll
    for (int c = 0; c < 64; ++c) {
      int g = c >> 3;
      float f = (bf2f(xr[c]) - stats[2 * g]) * stats[2 * g + 1] * n1g[c] + n1b[c];
      s_tok[t * 64 + c] = f2bf(f);
    }
  }
  __syncthreads();

  // QKV: out[tok, f] = sum_c tok[c] * wqkv[f][c] + bqkv[f]
  for (int tid = wave; tid < 192; tid += 8) {
    int mt = tid & 15, ft = tid >> 4;
    v8f acc = z8;
#pragma unroll
    for (int kt = 0; kt < 2; ++kt) {
      v16bf a = load_a(s_tok, 64, mt * 16, kt * 32, 64);
      v16bf b = load_bT(wqkv, 64, kt * 32, ft * 16, 64);
      acc = wmma_bf16(a, b, acc);
    }
    int sel = ft >> 2;  // 0=q 1=k 2=v (uniform per tile)
    unsigned short* dst = (sel == 0) ? s_q : ((sel == 1) ? s_k : s_v);
    int ch = (ft & 3) * 16 + (l & 15);
    float bias = bqkv[ft * 16 + (l & 15)];
#pragma unroll
    for (int r = 0; r < 8; ++r) {
      int tok = mt * 16 + r + hi;
      dst[tok * 64 + ch] = f2bf(acc[r] + bias);
    }
  }
  __syncthreads();

  // per (head, 16-query tile): scores over all 256 keys, softmax, P@V
  for (int id = wave; id < 64; id += 8) {
    int h = id >> 4, qt = id & 15;
    int hc = h * 16;
    v8f sacc[16];
#pragma unroll
    for (int nt = 0; nt < 16; ++nt) sacc[nt] = z8;
    v16bf aq = load_a(s_q + hc, 64, qt * 16, 0, 16);  // K padded 16->32 with zeros
#pragma unroll
    for (int nt = 0; nt < 16; ++nt) {
      v16bf b = load_bT(s_k + hc, 64, 0, nt * 16, 16);
      sacc[nt] = wmma_bf16(aq, b, sacc[nt]);
    }
    // softmax over 256 keys; rows live as (VGPR r, lane-half)
    const float scale = 0.25f;  // (C/HEADS)^-0.5 = 16^-0.5
    float rmax[8], rsum[8];
#pragma unroll
    for (int r = 0; r < 8; ++r) rmax[r] = -3.0e38f;
#pragma unroll
    for (int nt = 0; nt < 16; ++nt)
#pragma unroll
      for (int r = 0; r < 8; ++r) rmax[r] = fmaxf(rmax[r], sacc[nt][r]);
#pragma unroll
    for (int r = 0; r < 8; ++r) {
      for (int m = 1; m < 16; m <<= 1)
        rmax[r] = fmaxf(rmax[r], __shfl_xor(rmax[r], m, 32));
      rsum[r] = 0.f;
    }
#pragma unroll
    for (int nt = 0; nt < 16; ++nt)
#pragma unroll
      for (int r = 0; r < 8; ++r) {
        float e = __expf((sacc[nt][r] - rmax[r]) * scale);
        sacc[nt][r] = e;
        rsum[r] += e;
      }
#pragma unroll
    for (int r = 0; r < 8; ++r) {
      for (int m = 1; m < 16; m <<= 1) rsum[r] += __shfl_xor(rsum[r], m, 32);
      rsum[r] = 1.f / rsum[r];
    }
    // P @ V: stage P chunks through LDS to re-layout into A fragments
    unsigned short* pst = s_p[wave];
    v8f oacc = z8;
    for (int kc = 0; kc < 8; ++kc) {
#pragma unroll
      for (int half = 0; half < 2; ++half) {
        int f = 2 * kc + half;
#pragma unroll
        for (int r = 0; r < 8; ++r)
          pst[(r + hi) * 32 + half * 16 + (l & 15)] = f2bf(sacc[f][r] * rsum[r]);
      }
      v16bf a = load_a(pst, 32, 0, 0, 32);
      v16bf b = load_b(s_v + hc + kc * 32 * 64, 64, 0, 0, 32);
      oacc = wmma_bf16(a, b, oacc);
    }
#pragma unroll
    for (int r = 0; r < 8; ++r) {
      int tok = qt * 16 + r + hi;
      s_tok[tok * 64 + hc + (l & 15)] = f2bf(oacc[r]);
    }
  }
  __syncthreads();

  // proj + bias + residual scatter
  for (int tid = wave; tid < 64; tid += 8) {
    int mt = tid & 15, jt = tid >> 4;
    v8f acc = z8;
#pragma unroll
    for (int kt = 0; kt < 2; ++kt) {
      v16bf a = load_a(s_tok, 64, mt * 16, kt * 32, 64);
      v16bf b = load_bT(wproj, 64, kt * 32, jt * 16, 64);
      acc = wmma_bf16(a, b, acc);
    }
    int j = jt * 16 + (l & 15);
    float bias = bproj[j];
#pragma unroll
    for (int r = 0; r < 8; ++r) {
      int tok = mt * 16 + r + hi;
      int v = win_voxel(w, tok, sx, sy, sz);
      x[j * VTOT + v] += acc[r] + bias;
    }
  }
}

// ---------- fused FFN: x += f2 @ silu(f1 @ GN(x) + b1) + b2 ----------
__global__ __launch_bounds__(256) void k_ffn(
    float* __restrict__ x, const float* __restrict__ stats,
    const float* __restrict__ n2g, const float* __restrict__ n2b,
    const unsigned short* __restrict__ wf1, const float* __restrict__ b1,
    const unsigned short* __restrict__ wf2, const float* __restrict__ b2) {
  __shared__ unsigned short s_hid[8][128 * 16];
  int wave = threadIdx.x >> 5;
  int l = threadIdx.x & 31;
  int col0 = (blockIdx.x * 8 + wave) * 16;
  int n = l & 15, kh = (l >> 4) * 16, hi = (l >> 4) * 8;
  int v = col0 + n;
  const v8f z8 = ZERO8;

  // B fragments = GN(x) on the fly (64 channels = 2 K-steps)
  v16bf bfr[2];
#pragma unroll
  for (int kt = 0; kt < 2; ++kt) {
#pragma unroll
    for (int j = 0; j < 8; ++j) {
#pragma unroll
      for (int e = 0; e < 2; ++e) {
        int k = kt * 32 + kh + 2 * j + e;
        int g = k >> 3;
        float f = (x[k * VTOT + v] - stats[2 * g]) * stats[2 * g + 1] * n2g[k] + n2b[k];
        bfr[kt][2 * j + e] = bfbits(f2bf(f));
      }
    }
  }
  // phase 1: hid[128,16] = silu(wf1 @ B + b1)
  unsigned short* hp = s_hid[wave];
#pragma unroll
  for (int mt = 0; mt < 8; ++mt) {
    v8f acc = z8;
#pragma unroll
    for (int kt = 0; kt < 2; ++kt) {
      v16bf a = load_a(wf1, 64, mt * 16, kt * 32, 64);
      acc = wmma_bf16(a, bfr[kt], acc);
    }
#pragma unroll
    for (int r = 0; r < 8; ++r) {
      int m = mt * 16 + r + hi;
      hp[m * 16 + n] = f2bf(silu_f(acc[r] + b1[m]));
    }
  }
  // phase 2: x[64,16] += wf2 @ hid + b2
#pragma unroll
  for (int mt = 0; mt < 4; ++mt) {
    v8f acc = z8;
#pragma unroll
    for (int kt = 0; kt < 4; ++kt) {
      v16bf a = load_a(wf2, 128, mt * 16, kt * 32, 128);
      v16bf b = load_b(hp, 16, kt * 32, 0, 128);
      acc = wmma_bf16(a, b, acc);
    }
#pragma unroll
    for (int r = 0; r < 8; ++r) {
      int c = mt * 16 + r + hi;
      x[c * VTOT + v] += acc[r] + b2[c];
    }
  }
}

// ---------- head GEMM: out[32,V] = head_w @ x + b (bf16 activations) ----------
__global__ __launch_bounds__(256) void k_head(
    const unsigned short* __restrict__ xt, const unsigned short* __restrict__ wh,
    const float* __restrict__ bh, float* __restrict__ out) {
  int wave = threadIdx.x >> 5;
  int l = threadIdx.x & 31;
  int col0 = (blockIdx.x * 8 + wave) * 16;
  int n = l & 15, kh = (l >> 4) * 16, hi = (l >> 4) * 8;
  int v = col0 + n;
  const unsigned short* xr = xt + v * 64;
  v8f acc[2] = {ZERO8, ZERO8};
#pragma unroll
  for (int kt = 0; kt < 2; ++kt) {
    frag16 b;
#pragma unroll
    for (int j = 0; j < 8; ++j) {
      int k = kt * 32 + kh + 2 * j;
      b.d[j] = *(const u32*)(xr + k);
    }
#pragma unroll
    for (int mt = 0; mt < 2; ++mt) {
      v16bf a = load_a(wh, 64, mt * 16, kt * 32, 64);
      acc[mt] = wmma_bf16(a, b.v, acc[mt]);
    }
  }
#pragma unroll
  for (int mt = 0; mt < 2; ++mt)
#pragma unroll
    for (int r = 0; r < 8; ++r) {
      int m = mt * 16 + r + hi;
      out[m * VTOT + v] = acc[mt][r] + bh[m];
    }
}

// ---------------------------------------------------------------------------
extern "C" void kernel_launch(void* const* d_in, const int* in_sizes, int n_in,
                              void* d_out, int out_size, void* d_ws, size_t ws_size,
                              hipStream_t stream) {
  (void)in_sizes; (void)n_in; (void)out_size; (void)ws_size;
  const float* h_warp = (const float*)d_in[0];
  const float* fpa    = (const float*)d_in[1];
  const float* fc     = (const float*)d_in[2];
  const float* dt     = (const float*)d_in[3];
  const float* stem_w = (const float*)d_in[4];
  const float* stem_g = (const float*)d_in[5];
  const float* stem_b = (const float*)d_in[6];
  const float* rb1_w  = (const float*)d_in[7];
  const float* rb1_g  = (const float*)d_in[8];
  const float* rb1_b  = (const float*)d_in[9];
  const float* rb2_w  = (const float*)d_in[34];
  const float* rb2_g  = (const float*)d_in[35];
  const float* rb2_b  = (const float*)d_in[36];
  const float* head_w = (const float*)d_in[37];
  const float* head_b = (const float*)d_in[38];

  // workspace layout
  float* xbuf = (float*)d_ws;
  float* ybuf = xbuf + (size_t)64 * VTOT;
  float* partial = ybuf + (size_t)64 * VTOT;
  float* stats = partial + 8 * 64 * 2;
  unsigned short* xt = (unsigned short*)(stats + 16);        // [V][64] bf16
  unsigned short* wp = xt + (size_t)64 * VTOT;
  unsigned short* w_stem = wp;  wp += 64 * 128;              // padded lda=128
  unsigned short* w_conv1 = wp; wp += 27 * 64 * 64;
  unsigned short* w_conv2 = wp; wp += 27 * 64 * 64;
  unsigned short *w_qkv[2], *w_proj[2], *w_f1[2], *w_f2[2];
  for (int i = 0; i < 2; ++i) {
    w_qkv[i] = wp;  wp += 192 * 64;
    w_proj[i] = wp; wp += 64 * 64;
    w_f1[i] = wp;   wp += 128 * 64;
    w_f2[i] = wp;   wp += 64 * 128;
  }
  unsigned short* w_head = wp; wp += 32 * 64;

  auto cvt = [&](const float* s, unsigned short* d, int n) {
    k_cvt_bf16<<<(n + 255) / 256, 256, 0, stream>>>(s, d, n);
  };
  k_cvt_stem<<<(64 * 128 + 255) / 256, 256, 0, stream>>>(stem_w, w_stem);
  k_repack_conv<<<(27 * 64 * 64 + 255) / 256, 256, 0, stream>>>(rb1_w, w_conv1);
  k_repack_conv<<<(27 * 64 * 64 + 255) / 256, 256, 0, stream>>>(rb2_w, w_conv2);
  for (int i = 0; i < 2; ++i) {
    int base = 10 + 12 * i;
    cvt((const float*)d_in[base + 2], w_qkv[i], 192 * 64);
    cvt((const float*)d_in[base + 4], w_proj[i], 64 * 64);
    cvt((const float*)d_in[base + 8], w_f1[i], 128 * 64);
    cvt((const float*)d_in[base + 10], w_f2[i], 64 * 128);
  }
  cvt(head_w, w_head, 32 * 64);

  auto gnstats = [&](const float* y) {
    k_gn_part<<<dim3(64, 8), 256, 0, stream>>>(y, partial);
    k_gn_final<<<1, 32, 0, stream>>>(partial, stats, 64);
  };
  auto x2bf = [&]() {
    k_x2bf<<<VTOT / 64, 256, 0, stream>>>(xbuf, xt);
  };
  const int EW = (64 * VTOT + 255) / 256;

  // stem
  k_stem<<<VTOT / 128, 256, 0, stream>>>(h_warp, fpa, fc, dt, w_stem, ybuf);
  gnstats(ybuf);
  k_gn_act<<<EW, 256, 0, stream>>>(ybuf, (const float*)nullptr, stem_g, stem_b, stats, xbuf);
  x2bf();

  // res-dilated block 1 (d=1)
  k_conv<<<128 * 128 / 4, 128, 0, stream>>>(xt, w_conv1, ybuf, 1);
  gnstats(ybuf);
  k_gn_act<<<EW, 256, 0, stream>>>(ybuf, xbuf, rb1_g, rb1_b, stats, xbuf);
  x2bf();

  // attention blocks
  const int shifts[2][3] = {{0, 0, 0}, {4, 4, 0}};
  for (int i = 0; i < 2; ++i) {
    int base = 10 + 12 * i;
    const float* n1g = (const float*)d_in[base + 0];
    const float* n1b = (const float*)d_in[base + 1];
    const float* qb  = (const float*)d_in[base + 3];
    const float* pb  = (const float*)d_in[base + 5];
    const float* n2g = (const float*)d_in[base + 6];
    const float* n2b = (const float*)d_in[base + 7];
    const float* f1b = (const float*)d_in[base + 9];
    const float* f2b = (const float*)d_in[base + 11];
    gnstats(xbuf);
    k_attn<<<1024, 256, 0, stream>>>(xbuf, xt, stats, n1g, n1b, w_qkv[i], qb,
                                     w_proj[i], pb,
                                     shifts[i][0], shifts[i][1], shifts[i][2]);
    gnstats(xbuf);
    k_ffn<<<VTOT / 128, 256, 0, stream>>>(xbuf, stats, n2g, n2b, w_f1[i], f1b, w_f2[i], f2b);
    x2bf();  // refresh bf16 activations for next consumer (attn2 / conv2)
  }

  // res-dilated block 2 (d=2)
  k_conv<<<128 * 128 / 4, 128, 0, stream>>>(xt, w_conv2, ybuf, 2);
  gnstats(ybuf);
  k_gn_act<<<EW, 256, 0, stream>>>(ybuf, xbuf, rb2_g, rb2_b, stats, xbuf);
  x2bf();

  // head
  k_head<<<VTOT / 128, 256, 0, stream>>>(xt, w_head, head_b, (float*)d_out);
}
